// GEGNNLayer_55482387530475
// MI455X (gfx1250) — compile-verified, matrix operand order checked by reading, not run on hardware
//
#include <hip/hip_runtime.h>
#include <hip/hip_bf16.h>
#include <math.h>

// ---------------------------------------------------------------------------
// GEGNN layer for MI455X (gfx1250, wave32, WMMA, async-to-LDS DMA).
//   out = h + h@Wh1 + b_h                       (node kernel, WMMA)
//   out += scatter( silu(gather@W_a)@Wh2 )      (angle kernel, WMMA x2, atomics)
//   out += scatter( silu(gather@W_d)@Wh3 )      (dihedral kernel)
// Weights pre-packed into bf16 WMMA B-fragment layout in d_ws; h pre-converted
// to bf16 so edge-feature gathers run as GLOBAL_LOAD_ASYNC_TO_LDS_B128 DMA.
// Edge blocks process 32 edges (2 M-tiles per wave) so every B fragment load
// feeds two WMMAs.
// ---------------------------------------------------------------------------

typedef __attribute__((ext_vector_type(16))) __bf16 v16bf;
typedef __attribute__((ext_vector_type(4)))  __bf16 v4bf;
typedef __attribute__((ext_vector_type(8)))  float  v8f;

union AFrag { v16bf v; unsigned u[8]; };
union BFrag { v16bf v; uint4 q[2]; };

__device__ __forceinline__ float silu_f(float x) { return x / (1.0f + __expf(-x)); }

// LDS byte offset of a __shared__ object = low 32 bits of its flat address
// (ISA 10.2: LDS aperture maps addr[31:0] to the wave's LDS allocation).
__device__ __forceinline__ unsigned lds_off_u32(const void* p) {
  return (unsigned)(size_t)p;
}

// K index inside a 16x32 bf16 A fragment / 32x16 B fragment:
//   element e (0..15), lane-half hf (0/1):  K = e + (e&8) + hf*8
// (CDNA5 ISA 7.12.2, 16-bit A-matrix 16x32 table; B mirrors it with
//  lane%16 = column instead of row.)

// ---- pack [K x 128] row-major f32 weights -> bf16 fragment-major -----------
__global__ __launch_bounds__(256)
void pack_w_kernel(const float* __restrict__ W, __bf16* __restrict__ outp, int ktiles) {
  int i = blockIdx.x * 256 + threadIdx.x;
  int total = ktiles * 8 * 32 * 16;
  if (i >= total) return;
  int e    = i & 15;
  int lane = (i >> 4) & 31;
  int nt   = (i >> 9) & 7;
  int kt   = i >> 12;
  int hf   = lane >> 4;
  int K    = kt * 32 + e + (e & 8) + hf * 8;
  int col  = nt * 16 + (lane & 15);
  outp[i] = (__bf16)W[(size_t)K * 128 + col];
}

// ---- convert h (f32, row-major [N,128]) to bf16 once ----------------------
__global__ __launch_bounds__(256)
void h2bf_kernel(const float* __restrict__ h, __bf16* __restrict__ hbf, long total4) {
  long i = (long)blockIdx.x * 256 + threadIdx.x;
  if (i >= total4) return;
  float4 f = ((const float4*)h)[i];
  v4bf o;
  o[0] = (__bf16)f.x; o[1] = (__bf16)f.y; o[2] = (__bf16)f.z; o[3] = (__bf16)f.w;
  ((v4bf*)hbf)[i] = o;
}

// ===========================================================================
// shared tail: GEMM1 (32 x F, dual M-tile) + SiLU + message@Wh GEMM + scatter
// ===========================================================================
template<int SEGS>
__device__ __forceinline__
void edge_core(const __bf16* tA, __bf16* tM, const int* sIdx, const float* sVal,
               const float* __restrict__ W, const float* __restrict__ bvec,
               const __bf16* __restrict__ Wpk, const __bf16* __restrict__ Whpk,
               float* __restrict__ out, int E, long e0, int tid) {
  constexpr int F  = SEGS * 128;
  constexpr int KT = SEGS * 4;
  const int lane = tid & 31;
  const int wv   = tid >> 5;
  const int hf   = lane >> 4;
  const int col  = wv * 16 + (lane & 15);
  const int row  = lane & 15;

  // ---- GEMM 1: [32 x F] @ Wpk, one B fragment feeds two WMMAs ----
  v8f acc0 = {}, acc1 = {};
  for (int kt = 0; kt < KT; ++kt) {
    BFrag b;
    const uint4* bp = (const uint4*)(Wpk + ((size_t)(kt * 8 + wv) * 32 + lane) * 16);
    b.q[0] = bp[0]; b.q[1] = bp[1];
    if (kt + 1 < KT)
      __builtin_prefetch(Wpk + ((size_t)((kt + 1) * 8 + wv) * 32 + lane) * 16, 0, 3);
    AFrag a0, a1;
#pragma unroll
    for (int j = 0; j < 8; ++j) {
      int K0 = 2 * j + ((j & 4) ? 8 : 0) + hf * 8;
      a0.u[j] = *(const unsigned*)&tA[row * F + kt * 32 + K0];
      a1.u[j] = *(const unsigned*)&tA[(row + 16) * F + kt * 32 + K0];
    }
    acc0 = __builtin_amdgcn_wmma_f32_16x16x32_bf16(false, a0.v, false, b.v,
                                                   (short)0, acc0, false, false);
    acc1 = __builtin_amdgcn_wmma_f32_16x16x32_bf16(false, a1.v, false, b.v,
                                                   (short)0, acc1, false, false);
  }

  // rank-1 tail for the scalar feature (row F of W), bias, SiLU
  const float wtail = W[(size_t)F * 128 + col];
  const float bias  = bvec[col];
#pragma unroll
  for (int r = 0; r < 8; ++r) {
    int m0 = r + hf * 8, m1 = m0 + 16;
    float x0 = acc0[r] + sVal[m0] * wtail + bias;
    float x1 = acc1[r] + sVal[m1] * wtail + bias;
    acc0[r] = silu_f(x0);
    acc1[r] = silu_f(x1);
  }

  // round-trip message tile through LDS into A-fragment form (bf16)
#pragma unroll
  for (int r = 0; r < 8; ++r) {
    int m0 = r + hf * 8;
    tM[m0 * 128 + col]        = (__bf16)acc0[r];
    tM[(m0 + 16) * 128 + col] = (__bf16)acc1[r];
  }
  __syncthreads();

  // ---- GEMM 2: [32 x 128] message @ Wh slice ----
  v8f c0 = {}, c1 = {};
  for (int kt = 0; kt < 4; ++kt) {
    BFrag b;
    const uint4* bp = (const uint4*)(Whpk + ((size_t)(kt * 8 + wv) * 32 + lane) * 16);
    b.q[0] = bp[0]; b.q[1] = bp[1];
    AFrag a0, a1;
#pragma unroll
    for (int j = 0; j < 8; ++j) {
      int K0 = 2 * j + ((j & 4) ? 8 : 0) + hf * 8;
      a0.u[j] = *(const unsigned*)&tM[row * 128 + kt * 32 + K0];
      a1.u[j] = *(const unsigned*)&tM[(row + 16) * 128 + kt * 32 + K0];
    }
    c0 = __builtin_amdgcn_wmma_f32_16x16x32_bf16(false, a0.v, false, b.v,
                                                 (short)0, c0, false, false);
    c1 = __builtin_amdgcn_wmma_f32_16x16x32_bf16(false, a1.v, false, b.v,
                                                 (short)0, c1, false, false);
  }

  // scatter-add into out at center node idx[:,1] (hardware f32 atomics, L2)
  if (e0 + 32 <= (long)E) {            // uniform full-tile fast path
#pragma unroll
    for (int r = 0; r < 8; ++r) {
      int m0 = r + hf * 8, m1 = m0 + 16;
      unsafeAtomicAdd(&out[(size_t)sIdx[m0 * SEGS + 1] * 128 + col], c0[r]);
      unsafeAtomicAdd(&out[(size_t)sIdx[m1 * SEGS + 1] * 128 + col], c1[r]);
    }
  } else {
#pragma unroll
    for (int r = 0; r < 8; ++r) {
      int m0 = r + hf * 8, m1 = m0 + 16;
      if (e0 + m0 < E)
        unsafeAtomicAdd(&out[(size_t)sIdx[m0 * SEGS + 1] * 128 + col], c0[r]);
      if (e0 + m1 < E)
        unsafeAtomicAdd(&out[(size_t)sIdx[m1 * SEGS + 1] * 128 + col], c1[r]);
    }
  }
}

// ===========================================================================
// async variant: bf16 h, feature tile staged by GLOBAL_LOAD_ASYNC_TO_LDS_B128
// ===========================================================================
template<int SEGS>
__global__ __launch_bounds__(256)
void edge_msg_kernel_async(const __bf16* __restrict__ hbf,
                           const int*   __restrict__ idx,
                           const float* __restrict__ val,
                           const float* __restrict__ W,
                           const float* __restrict__ bvec,
                           const __bf16* __restrict__ Wpk,
                           const __bf16* __restrict__ Whpk,
                           float* __restrict__ out, int E) {
  constexpr int F = SEGS * 128;
  __shared__ __bf16 tA[32 * F];
  __shared__ __bf16 tM[32 * 128];
  __shared__ int    sIdx[32 * SEGS];
  __shared__ float  sVal[32];

  const int tid = threadIdx.x;
  const long e0 = (long)blockIdx.x * 32;

  if (tid < 32 * SEGS) {
    long e = e0 + tid / SEGS; if (e >= E) e = E - 1;
    sIdx[tid] = idx[e * SEGS + (tid % SEGS)];
  }
  if (tid < 32) {
    long e = e0 + tid; if (e >= E) e = E - 1;
    sVal[tid] = val[e];
  }
  __syncthreads();

  // DMA gather: 32 edges * SEGS rows * 256B, as 16B chunks, L2 -> LDS,
  // no VGPR round-trip (ASYNCcnt path, CDNA5 ISA).
  {
    constexpr int CHUNKS = 32 * SEGS * 16;           // 1536 (x3) / 2048 (x4)
    const unsigned ldsbase = lds_off_u32(tA);
    const unsigned long long gbase = (unsigned long long)(size_t)hbf;
#pragma unroll
    for (int it = 0; it < CHUNKS / 256; ++it) {
      int cid   = tid + it * 256;
      int edge  = cid / (SEGS * 16);
      int seg   = (cid >> 4) % SEGS;
      int chunk = cid & 15;
      int node  = sIdx[edge * SEGS + seg];
      unsigned long long ga = gbase + (unsigned long long)node * 256ull
                                    + (unsigned)(chunk * 16);
      unsigned la = ldsbase + (unsigned)(cid * 16);
      asm volatile("global_load_async_to_lds_b128 %0, %1, off"
                   :: "v"(la), "v"(ga) : "memory");
    }
    asm volatile("s_wait_asynccnt 0x0" ::: "memory");
  }
  __syncthreads();

  edge_core<SEGS>(tA, tM, sIdx, sVal, W, bvec, Wpk, Whpk, out, E, e0, tid);
}

// ===========================================================================
// sync fallback (f32 h gathered through VGPRs, converted on the fly)
// ===========================================================================
template<int SEGS>
__global__ __launch_bounds__(256)
void edge_msg_kernel_sync(const float* __restrict__ h,
                          const int*   __restrict__ idx,
                          const float* __restrict__ val,
                          const float* __restrict__ W,
                          const float* __restrict__ bvec,
                          const __bf16* __restrict__ Wpk,
                          const __bf16* __restrict__ Whpk,
                          float* __restrict__ out, int E) {
  constexpr int F = SEGS * 128;
  __shared__ __bf16 tA[32 * F];
  __shared__ __bf16 tM[32 * 128];
  __shared__ int    sIdx[32 * SEGS];
  __shared__ float  sVal[32];

  const int tid = threadIdx.x;
  const long e0 = (long)blockIdx.x * 32;

  if (tid < 32 * SEGS) {
    long e = e0 + tid / SEGS; if (e >= E) e = E - 1;
    sIdx[tid] = idx[e * SEGS + (tid % SEGS)];
  }
  if (tid < 32) {
    long e = e0 + tid; if (e >= E) e = E - 1;
    sVal[tid] = val[e];
  }
  __syncthreads();

  // stage 32 x F feature tile into LDS as bf16 (float4 loads, bf16x4 stores)
  for (int i = tid; i < 32 * (F / 4); i += 256) {
    int m = i / (F / 4), c = i % (F / 4);
    int f = c * 4;
    int node = sIdx[m * SEGS + (f >> 7)];
    float4 v = *(const float4*)&h[(size_t)node * 128 + (f & 127)];
    v4bf o;
    o[0] = (__bf16)v.x; o[1] = (__bf16)v.y; o[2] = (__bf16)v.z; o[3] = (__bf16)v.w;
    *(v4bf*)&tA[m * F + f] = o;
  }
  __syncthreads();

  edge_core<SEGS>(tA, tM, sIdx, sVal, W, bvec, Wpk, Whpk, out, E, e0, tid);
}

// ---- node kernel: out = h + h@Wh1 + b_h ------------------------------------
__global__ __launch_bounds__(256)
void node_kernel(const float* __restrict__ h,
                 const float* __restrict__ bvec,
                 const __bf16* __restrict__ Wh1pk,
                 float* __restrict__ out, int Nn) {
  __shared__ __bf16 tA[16 * 128];
  const int tid = threadIdx.x;
  const long r0 = (long)blockIdx.x * 16;

  for (int i = tid; i < 16 * 32; i += 256) {
    int m = i >> 5, c = (i & 31) * 4;
    long rrow = r0 + m; if (rrow >= Nn) rrow = Nn - 1;
    float4 v = *(const float4*)&h[rrow * 128 + c];
    v4bf o;
    o[0] = (__bf16)v.x; o[1] = (__bf16)v.y; o[2] = (__bf16)v.z; o[3] = (__bf16)v.w;
    *(v4bf*)&tA[m * 128 + c] = o;
  }
  __syncthreads();

  const int lane = tid & 31;
  const int wv   = tid >> 5;
  const int hf   = lane >> 4;
  const int col  = wv * 16 + (lane & 15);
  const int row  = lane & 15;

  v8f acc = {};
  for (int kt = 0; kt < 4; ++kt) {
    AFrag a;
#pragma unroll
    for (int j = 0; j < 8; ++j) {
      int K0 = 2 * j + ((j & 4) ? 8 : 0) + hf * 8;
      a.u[j] = *(const unsigned*)&tA[row * 128 + kt * 32 + K0];
    }
    BFrag b;
    const uint4* bp = (const uint4*)(Wh1pk + ((size_t)(kt * 8 + wv) * 32 + lane) * 16);
    b.q[0] = bp[0]; b.q[1] = bp[1];
    acc = __builtin_amdgcn_wmma_f32_16x16x32_bf16(false, a.v, false, b.v,
                                                  (short)0, acc, false, false);
  }

  const float bias = bvec[col];
  if (r0 + 16 <= (long)Nn) {           // uniform full-tile fast path
#pragma unroll
    for (int r = 0; r < 8; ++r) {
      size_t o = (size_t)(r0 + r + hf * 8) * 128 + col;
      out[o] = h[o] + acc[r] + bias;
    }
  } else {
#pragma unroll
    for (int r = 0; r < 8; ++r) {
      long rrow = r0 + r + hf * 8;
      if (rrow < Nn) {
        size_t o = (size_t)rrow * 128 + col;
        out[o] = h[o] + acc[r] + bias;
      }
    }
  }
}

// ---------------------------------------------------------------------------
extern "C" void kernel_launch(void* const* d_in, const int* in_sizes, int n_in,
                              void* d_out, int out_size, void* d_ws, size_t ws_size,
                              hipStream_t stream) {
  const float* h     = (const float*)d_in[0];
  const int*   a_idx = (const int*)  d_in[1];
  const float* a_val = (const float*)d_in[2];
  const int*   d_idx = (const int*)  d_in[3];
  const float* d_val = (const float*)d_in[4];
  const float* W_a   = (const float*)d_in[5];
  const float* b_a   = (const float*)d_in[6];
  const float* W_d   = (const float*)d_in[7];
  const float* b_d   = (const float*)d_in[8];
  const float* W_h   = (const float*)d_in[9];
  const float* b_h   = (const float*)d_in[10];
  float* out = (float*)d_out;

  const int Nn  = in_sizes[0] / 128;
  const int E_a = in_sizes[1] / 3;
  const int E_d = in_sizes[3] / 4;

  // workspace layout: packed bf16 weight fragments, then optional bf16 h
  const size_t elWa = (size_t)12 * 8 * 32 * 16;
  const size_t elWd = (size_t)16 * 8 * 32 * 16;
  const size_t elWh = (size_t)4  * 8 * 32 * 16;
  __bf16* Wa_pk  = (__bf16*)d_ws;
  __bf16* Wd_pk  = Wa_pk  + elWa;
  __bf16* Wh1_pk = Wd_pk  + elWd;
  __bf16* Wh2_pk = Wh1_pk + elWh;
  __bf16* Wh3_pk = Wh2_pk + elWh;
  __bf16* hbf    = Wh3_pk + elWh;
  const size_t wBytes    = (elWa + elWd + 3 * elWh) * sizeof(__bf16);
  const size_t needAsync = wBytes + (size_t)Nn * 128 * sizeof(__bf16);
  const bool useAsync = (ws_size >= needAsync);

  pack_w_kernel<<<(int)((elWa + 255) / 256), 256, 0, stream>>>(W_a, Wa_pk, 12);
  pack_w_kernel<<<(int)((elWd + 255) / 256), 256, 0, stream>>>(W_d, Wd_pk, 16);
  pack_w_kernel<<<(int)((elWh + 255) / 256), 256, 0, stream>>>(W_h,             Wh1_pk, 4);
  pack_w_kernel<<<(int)((elWh + 255) / 256), 256, 0, stream>>>(W_h + 128 * 128, Wh2_pk, 4);
  pack_w_kernel<<<(int)((elWh + 255) / 256), 256, 0, stream>>>(W_h + 256 * 128, Wh3_pk, 4);

  // out = h + h@Wh1 + b_h   (must run before atomic scatters)
  node_kernel<<<(Nn + 15) / 16, 256, 0, stream>>>(h, b_h, Wh1_pk, out, Nn);

  if (useAsync) {
    const long t4 = (long)Nn * 32;                  // N*128/4
    h2bf_kernel<<<(int)((t4 + 255) / 256), 256, 0, stream>>>(h, hbf, t4);
    edge_msg_kernel_async<3><<<(E_a + 31) / 32, 256, 0, stream>>>(
        hbf, a_idx, a_val, W_a, b_a, Wa_pk, Wh2_pk, out, E_a);
    edge_msg_kernel_async<4><<<(E_d + 31) / 32, 256, 0, stream>>>(
        hbf, d_idx, d_val, W_d, b_d, Wd_pk, Wh3_pk, out, E_d);
  } else {
    edge_msg_kernel_sync<3><<<(E_a + 31) / 32, 256, 0, stream>>>(
        h, a_idx, a_val, W_a, b_a, Wa_pk, Wh2_pk, out, E_a);
    edge_msg_kernel_sync<4><<<(E_d + 31) / 32, 256, 0, stream>>>(
        h, d_idx, d_val, W_d, b_d, Wd_pk, Wh3_pk, out, E_d);
  }
}
